// MemoryMoCo_10359461118227
// MI455X (gfx1250) — matrix operand order repaired and checked by reading, not compiled, not run on hardware
//
#include <hip/hip_runtime.h>

// ---------------------------------------------------------------------------
// MemoryMoCo forward for MI455X (gfx1250, wave32, WMMA).
// Roofline: 25.8 GFLOP vs ~530 MB traffic. bf16 WMMA (16x16x32) keeps the
// kernel HBM-bound (~23 us @ 23.3 TB/s); fp32 WMMA (16x16x4) would be
// compute-bound (~80 us). fp32->bf16 conversion happens in registers during
// the load, so the precision downgrade costs zero extra bandwidth.
//
// Round-2 changes:
//  * addrspace(1) casts so loads lower to global_load_b128 (flat loads were
//    incrementing DScnt and occupying the LDS path for no reason).
//  * non-temporal stores for the 403 MB out0 stream so it does not evict the
//    128 MB of memory banks from the 192 MB L2 (M rows have 16x tile reuse).
//  * each wave keeps all 4 K-step A fragments in registers and sweeps 4
//    N-tiles, reloading only B (4x less A load traffic, 16 WMMAs per wave).
// ---------------------------------------------------------------------------

typedef __attribute__((ext_vector_type(16))) __bf16 v16bf;
typedef __attribute__((ext_vector_type(8)))  float  v8f;

typedef __attribute__((address_space(1))) const float* gcf_t;   // global const
typedef __attribute__((address_space(1)))       float* gf_t;    // global

__constant__ int g_CI[6] = {0, 0, 1, 1, 2, 3};
__constant__ int g_CJ[6] = {1, 3, 0, 2, 1, 0};

static constexpr int   BN      = 256;      // batch
static constexpr int   QN      = 65536;    // memory slots
static constexpr int   DN      = 128;      // feature dim (= K)
static constexpr float TEMPINV = 1.0f / 0.07f;
static constexpr long long OUT0_LD = (long long)(QN + 1);   // 65537
static constexpr long long OUT0_SZ = 6LL * BN * OUT0_LD;

// Load the 4 K-step A fragments for this lane (ISA 7.12.2 16-bit A 16x32
// layout: lane L holds row M = L&15; K-halves {0..7,16..23} for lanes 0-15,
// {8..15,24..31} for lanes 16-31).
__device__ __forceinline__ void load_a_frags(gcf_t arow, int lane, v16bf af[4]) {
  const int aSel = (lane >> 4) << 3;   // 0 or 8
#pragma unroll
  for (int ks = 0; ks < 4; ++ks) {
    const int k0 = ks * 32;
#pragma unroll
    for (int t = 0; t < 8; ++t) {
      af[ks][t]     = (__bf16)arow[k0 + aSel + t];
      af[ks][t + 8] = (__bf16)arow[k0 + 16 + aSel + t];
    }
  }
}

// B fragment (32x16 bf16): lane L holds col N = L&15; K-halves
// {0..7,16..23} for lanes 0-15, {8..15,24..31} for lanes 16-31 (same as A)
// from the K-contiguous B-side memory row. Accumulate 4 WMMAs over K=128.
__device__ __forceinline__ v8f wmma_tile_128(const v16bf af[4], gcf_t brow,
                                             int lane) {
  const int bSel = (lane >> 4) << 3;   // 0 or 8
  v8f acc = {0.f, 0.f, 0.f, 0.f, 0.f, 0.f, 0.f, 0.f};
#pragma unroll
  for (int ks = 0; ks < 4; ++ks) {
    const int k0 = ks * 32;
    v16bf b;
#pragma unroll
    for (int t = 0; t < 16; ++t) b[t] = (__bf16)brow[k0 + bSel + (t & 7) + ((t >> 3) << 4)];
    acc = __builtin_amdgcn_wmma_f32_16x16x32_bf16(
        /*neg_a=*/false, af[ks], /*neg_b=*/false, b,
        /*c_mod=*/(short)0, acc, /*reuse_a=*/false, /*reuse_b=*/false);
  }
  return acc;
}

// ---------------------------------------------------------------------------
// Row L2-norms: one wave per row across 4 arrays of R rows x 128 cols.
// ---------------------------------------------------------------------------
__global__ __launch_bounds__(256) void norms_kernel(
    const float* __restrict__ a0, const float* __restrict__ a1,
    const float* __restrict__ a2, const float* __restrict__ a3,
    float* __restrict__ out, int R) {
  const int lane = threadIdx.x & 31;
  const int gw   = blockIdx.x * (blockDim.x >> 5) + (threadIdx.x >> 5);
  if (gw >= 4 * R) return;
  const int v = gw / R, r = gw - v * R;
  const float* as[4] = {a0, a1, a2, a3};
  gcf_t row = (gcf_t)(as[v] + (size_t)r * DN);
  float s = 0.f;
#pragma unroll
  for (int t = lane; t < DN; t += 32) { float f = row[t]; s += f * f; }
#pragma unroll
  for (int off = 16; off >= 1; off >>= 1) s += __shfl_xor(s, off, 32);
  if (lane == 0) ((gf_t)out)[gw] = sqrtf(s);
}

// ---------------------------------------------------------------------------
// out0[:, :, 0] = exp(dot(X[i][b], X[j][b]) / (Xn_i * Xn_j) / T). 1 wave each.
// ---------------------------------------------------------------------------
__global__ __launch_bounds__(256) void pos_kernel(
    const float* __restrict__ x0, const float* __restrict__ x1,
    const float* __restrict__ x2, const float* __restrict__ x3,
    const float* __restrict__ xn, float* __restrict__ out0) {
  const int lane = threadIdx.x & 31;
  const int gw   = blockIdx.x * 8 + (threadIdx.x >> 5);   // 0..1535
  const int c = gw >> 8, b = gw & 255;
  const int i = g_CI[c], j = g_CJ[c];
  const float* xs[4] = {x0, x1, x2, x3};
  gcf_t pa = (gcf_t)(xs[i] + (size_t)b * DN);
  gcf_t pb = (gcf_t)(xs[j] + (size_t)b * DN);
  float s = 0.f;
#pragma unroll
  for (int t = lane; t < DN; t += 32) s += pa[t] * pb[t];
#pragma unroll
  for (int off = 16; off >= 1; off >>= 1) s += __shfl_xor(s, off, 32);
  if (lane == 0) {
    const float denom = ((gcf_t)xn)[i * BN + b] * ((gcf_t)xn)[j * BN + b];
    ((gf_t)out0)[(long long)c * BN * OUT0_LD + (long long)b * OUT0_LD] =
        __expf(s / denom * TEMPINV);
  }
}

// ---------------------------------------------------------------------------
// out0[:, :, 1:] = exp((X[i] @ M[j]^T) / (Xn_i x Mn_j) / T).
// grid (Q/512, B/16, 6); 8 waves/block; each wave: one 16-row stripe, 4
// N-tiles (64 cols), A fragments resident in VGPRs across the N sweep.
// ---------------------------------------------------------------------------
__global__ __launch_bounds__(256) void gemm_out0_kernel(
    const float* __restrict__ x0, const float* __restrict__ x1,
    const float* __restrict__ x2, const float* __restrict__ x3,
    const float* __restrict__ m0, const float* __restrict__ m1,
    const float* __restrict__ m2, const float* __restrict__ m3,
    const float* __restrict__ xn, const float* __restrict__ mn,
    float* __restrict__ out0) {
  const float* xs[4] = {x0, x1, x2, x3};
  const float* ms[4] = {m0, m1, m2, m3};
  const int c = blockIdx.z;
  const int i = g_CI[c], j = g_CJ[c];
  const int lane   = threadIdx.x & 31;
  const int wave   = threadIdx.x >> 5;
  const int mbase  = blockIdx.y << 4;
  const int nbase0 = (blockIdx.x * 8 + wave) << 6;   // 64 cols per wave
  const int r16    = lane & 15;

  v16bf af[4];
  load_a_frags((gcf_t)(xs[i] + (size_t)(mbase + r16) * DN), lane, af);

  // epilogue row norms, loaded once per wave
  const int rbase = mbase + ((lane >> 4) << 3);
  float xnr[8];
#pragma unroll
  for (int g = 0; g < 8; ++g) xnr[g] = ((gcf_t)xn)[i * BN + rbase + g];

  const long long cbase = (long long)c * BN * OUT0_LD;
#pragma unroll
  for (int nt = 0; nt < 4; ++nt) {
    const int nbase = nbase0 + (nt << 4);
    gcf_t brow = (gcf_t)(ms[j] + (size_t)(nbase + r16) * DN);
    v8f acc = wmma_tile_128(af, brow, lane);

    const int   col = nbase + r16;
    const float mnv = ((gcf_t)mn)[(size_t)j * QN + col];
    const long long obase = cbase + 1 + col;
#pragma unroll
    for (int g = 0; g < 8; ++g) {
      const float v = __expf(acc[g] / (xnr[g] * mnv) * TEMPINV);
      __builtin_nontemporal_store(
          v, (gf_t)(out0 + obase + (long long)(rbase + g) * OUT0_LD));
    }
  }
}

// ---------------------------------------------------------------------------
// T[c] = exp((X[i] @ X[j]^T) / (Xn_i x Xn_j) / T), pre-gather. grid (1,16,6);
// each wave sweeps 4 N-tiles -> 8 waves cover all 256 cols.
// ---------------------------------------------------------------------------
__global__ __launch_bounds__(256) void gemm_out1_kernel(
    const float* __restrict__ x0, const float* __restrict__ x1,
    const float* __restrict__ x2, const float* __restrict__ x3,
    const float* __restrict__ xn, float* __restrict__ T) {
  const float* xs[4] = {x0, x1, x2, x3};
  const int c = blockIdx.z;
  const int i = g_CI[c], j = g_CJ[c];
  const int lane   = threadIdx.x & 31;
  const int wave   = threadIdx.x >> 5;
  const int mbase  = blockIdx.y << 4;
  const int nbase0 = wave << 6;
  const int r16    = lane & 15;

  v16bf af[4];
  load_a_frags((gcf_t)(xs[i] + (size_t)(mbase + r16) * DN), lane, af);

  const int rbase = mbase + ((lane >> 4) << 3);
  float xnr[8];
#pragma unroll
  for (int g = 0; g < 8; ++g) xnr[g] = ((gcf_t)xn)[i * BN + rbase + g];

#pragma unroll
  for (int nt = 0; nt < 4; ++nt) {
    const int nbase = nbase0 + (nt << 4);
    gcf_t brow = (gcf_t)(xs[j] + (size_t)(nbase + r16) * DN);
    v8f acc = wmma_tile_128(af, brow, lane);

    const int   col = nbase + r16;
    const float nj  = ((gcf_t)xn)[j * BN + col];
#pragma unroll
    for (int g = 0; g < 8; ++g) {
      ((gf_t)T)[((long long)c << 16) + ((rbase + g) << 8) + col] =
          __expf(acc[g] / (xnr[g] * nj) * TEMPINV);
    }
  }
}

// out1[c,k,l] = T[c,k,(k+l) % 256]
__global__ __launch_bounds__(256) void gather_out1_kernel(
    const float* __restrict__ T, float* __restrict__ out1) {
  const int idx = blockIdx.x * blockDim.x + threadIdx.x;   // < 6*65536
  const int c = idx >> 16;
  const int rem = idx & 65535;
  const int k = rem >> 8, l = rem & 255;
  const float v = ((gcf_t)T)[((long long)c << 16) + (k << 8) + ((k + l) & 255)];
  __builtin_nontemporal_store(v, (gf_t)(out1 + idx));
}

// ---------------------------------------------------------------------------
extern "C" void kernel_launch(void* const* d_in, const int* in_sizes, int n_in,
                              void* d_out, int out_size, void* d_ws, size_t ws_size,
                              hipStream_t stream) {
  // setup_inputs() dict order: x0, m0, x1, m1, x2, m2, x3, m3
  const float* x0 = (const float*)d_in[0];
  const float* m0 = (const float*)d_in[1];
  const float* x1 = (const float*)d_in[2];
  const float* m1 = (const float*)d_in[3];
  const float* x2 = (const float*)d_in[4];
  const float* m2 = (const float*)d_in[5];
  const float* x3 = (const float*)d_in[6];
  const float* m3 = (const float*)d_in[7];

  float* out0 = (float*)d_out;
  float* out1 = out0 + OUT0_SZ;

  // workspace: Xn[4*256] | Mn[4*65536] | T[6*256*256]  (~2.6 MB)
  float* xn = (float*)d_ws;
  float* mn = xn + 4 * BN;
  float* T  = mn + 4 * QN;

  norms_kernel<<<(4 * BN + 7) / 8, 256, 0, stream>>>(x0, x1, x2, x3, xn, BN);
  norms_kernel<<<(4 * QN + 7) / 8, 256, 0, stream>>>(m0, m1, m2, m3, mn, QN);
  pos_kernel<<<(6 * BN) / 8, 256, 0, stream>>>(x0, x1, x2, x3, xn, out0);
  gemm_out0_kernel<<<dim3(QN / 512, BN / 16, 6), 256, 0, stream>>>(
      x0, x1, x2, x3, m0, m1, m2, m3, xn, mn, out0);
  gemm_out1_kernel<<<dim3(1, BN / 16, 6), 256, 0, stream>>>(
      x0, x1, x2, x3, xn, T);
  gather_out1_kernel<<<(6 * BN * BN) / 256, 256, 0, stream>>>(T, out1);
}
